// ManifoldNet_63385127354809
// MI455X (gfx1250) — compile-verified, matrix operand order checked by reading, not run on hardware
//
#include <hip/hip_runtime.h>
#include <hip/hip_bf16.h>
#include <math.h>

typedef __attribute__((ext_vector_type(16))) _Float16 v16h;
typedef __attribute__((ext_vector_type(8)))  float    v8f;

#define EPSF 1e-8f

union V16H { v16h h; unsigned u[8]; };
union F4U  { float4 v; float f[4]; };

__device__ inline unsigned pack2h(float lo, float hi) {
  union { _Float16 h[2]; unsigned u; } p;
  p.h[0] = (_Float16)lo; p.h[1] = (_Float16)hi;
  return p.u;
}
__device__ inline unsigned short f2h(float v) {
  union { _Float16 h; unsigned short s; } u; u.h = (_Float16)v; return u.s;
}
__device__ inline v16h load_b16(const unsigned* p) {
  V16H r;
  uint4 q0 = ((const uint4*)p)[0];
  uint4 q1 = ((const uint4*)p)[1];
  r.u[0]=q0.x; r.u[1]=q0.y; r.u[2]=q0.z; r.u[3]=q0.w;
  r.u[4]=q1.x; r.u[5]=q1.y; r.u[6]=q1.z; r.u[7]=q1.w;
  return r.h;
}

// ---------------------------------------------------------------------------
// K0: softmax(w, axis=0) and pack B-matrix tiles in WMMA 16x16x32 f16 per-lane
// layout: B[(chunk*T + ntile)*32 + lane][j] = pack(wsm[K0], wsm[K0+1]),
//   K0 = 32*chunk + (j<4 ? 2j : 16+2(j-4)) + 8*(lane>=16), N = 16*ntile + lane%16
// ---------------------------------------------------------------------------
__global__ void k0_prep(const float* __restrict__ w1, const float* __restrict__ w2,
                        unsigned* __restrict__ B1, unsigned* __restrict__ B2) {
  int t = threadIdx.x;
  if (t < 32) {                       // w1: [96,30], columns 0..29 (+pad 30,31)
    int n = t;
    float mx = -3.4e38f, sum = 0.f;
    if (n < 30) {
      for (int k = 0; k < 96; ++k) mx = fmaxf(mx, w1[k*30 + n]);
      for (int k = 0; k < 96; ++k) sum += __expf(w1[k*30 + n] - mx);
    }
    int tile = n >> 4, ln = n & 15;
    for (int cc = 0; cc < 3; ++cc)
      for (int hf = 0; hf < 2; ++hf) {
        int L = ln + 16*hf;
        for (int j = 0; j < 8; ++j) {
          int k0 = 32*cc + ((j < 4) ? 2*j : 16 + 2*(j-4)) + 8*hf;
          float lo = 0.f, hi = 0.f;
          if (n < 30) {
            lo = __expf(w1[k0*30 + n] - mx) / sum;
            hi = __expf(w1[(k0+1)*30 + n] - mx) / sum;
          }
          B1[((size_t)(cc*2 + tile)*32 + L)*8 + j] = pack2h(lo, hi);
        }
      }
  } else if (t < 80) {                // w2: [960,40], columns 0..39 (+pad 40..47)
    int n = t - 32;
    float mx = -3.4e38f, sum = 0.f;
    if (n < 40) {
      for (int k = 0; k < 960; ++k) mx = fmaxf(mx, w2[k*40 + n]);
      for (int k = 0; k < 960; ++k) sum += __expf(w2[k*40 + n] - mx);
    }
    int tile = n >> 4, ln = n & 15;
    for (int cc = 0; cc < 30; ++cc)
      for (int hf = 0; hf < 2; ++hf) {
        int L = ln + 16*hf;
        for (int j = 0; j < 8; ++j) {
          int k0 = 32*cc + ((j < 4) ? 2*j : 16 + 2*(j-4)) + 8*hf;
          float lo = 0.f, hi = 0.f;
          if (n < 40) {
            lo = __expf(w2[k0*40 + n] - mx) / sum;
            hi = __expf(w2[(k0+1)*40 + n] - mx) / sum;
          }
          B2[((size_t)(cc*3 + tile)*32 + L)*8 + j] = pack2h(lo, hi);
        }
      }
  }
}

// ---------------------------------------------------------------------------
// K1: KNN. One block per batch; whole X_b (512x48 = 96KB, fits the 320KB LDS)
// tiled in LDS, one thread per query, streaming top-32 by replace-max + sort.
// ---------------------------------------------------------------------------
__global__ __launch_bounds__(512) void k1_knn(const float* __restrict__ x,
                                              int* __restrict__ idxout) {
  extern __shared__ char dynsmem[];
  float* xs = (float*)dynsmem;                    // 512*48 floats
  int b = blockIdx.x, q = threadIdx.x;
  const float* xb = x + (size_t)b * 512 * 48;
  for (int i = threadIdx.x; i < 512*48/4; i += 512)
    ((float4*)xs)[i] = ((const float4*)xb)[i];
  __syncthreads();

  float f[48];
  #pragma unroll
  for (int i = 0; i < 48; ++i) f[i] = xs[q*48 + i];

  float bd[32]; int bi[32];
  #pragma unroll
  for (int j = 0; j < 32; ++j) { bd[j] = 3.4e38f; bi[j] = 0; }
  float maxd = 3.4e38f; int maxj = 0;

  for (int m = 0; m < 512; ++m) {
    float d = 0.f;
    #pragma unroll
    for (int i = 0; i < 48; ++i) { float t = f[i] - xs[m*48 + i]; d = fmaf(t, t, d); }
    if (d < maxd) {
      bd[maxj] = d; bi[maxj] = m;
      float nm = -1.f; int nj = 0;
      #pragma unroll
      for (int j = 0; j < 32; ++j) if (bd[j] > nm) { nm = bd[j]; nj = j; }
      maxd = nm; maxj = nj;
    }
  }
  // ascending (distance, index) like top_k of -d
  for (int a = 0; a < 32; ++a) {
    int mj = a;
    for (int j = a+1; j < 32; ++j)
      if (bd[j] < bd[mj] || (bd[j] == bd[mj] && bi[j] < bi[mj])) mj = j;
    float td = bd[a]; bd[a] = bd[mj]; bd[mj] = td;
    int   ti = bi[a]; bi[a] = bi[mj]; bi[mj] = ti;
    idxout[((size_t)b*512 + q)*32 + a] = bi[a];
  }
}

// ---------------------------------------------------------------------------
// K2: wfm1 + nonlinear. One wave per point, 8 waves/block.
// Gather 32 neighbors (48 f32 each) -> f16 in LDS as [m][d], m = kk*3+c.
// y[16x30] = A[16x96] x softmax(w1)[96x30] via 3x2 v_wmma_f32_16x16x32_f16.
// Double sphere-normalize over D, store fm1 as f16 column pairs [pair*16+d].
// ---------------------------------------------------------------------------
__global__ __launch_bounds__(256) void k2_wfm1(const float* __restrict__ x,
                                               const int* __restrict__ idx,
                                               const unsigned* __restrict__ B1,
                                               unsigned* __restrict__ fm1h) {
  __shared__ unsigned short a16[8 * 1536];
  int wave = threadIdx.x >> 5, lane = threadIdx.x & 31;
  int point = blockIdx.x * 8 + wave;
  int b = point >> 9;
  unsigned short* lds = a16 + wave * 1536;

  // stage: lane kk copies its neighbor's 48 features (f32 -> f16)
  int nkk = idx[(size_t)point*32 + lane];
  const float* src = x + (size_t)(b*512 + nkk) * 48;
  #pragma unroll
  for (int i = 0; i < 12; ++i) {
    F4U v; v.v = ((const float4*)src)[i];
    #pragma unroll
    for (int s = 0; s < 4; ++s) {
      int e = i*4 + s, d = e/3, c = e%3;
      lds[(lane*3 + c)*16 + d] = f2h(v.f[s]);
    }
  }
  __syncthreads();

  int hf = lane >> 4, ln = lane & 15;
  v8f acc0 = {}, acc1 = {};
  #pragma unroll
  for (int cc = 0; cc < 3; ++cc) {
    V16H a;
    #pragma unroll
    for (int j = 0; j < 8; ++j) {
      int k0 = 32*cc + ((j < 4) ? 2*j : 16 + 2*(j-4)) + 8*hf;
      unsigned lo = lds[k0*16 + ln];
      unsigned hi = lds[(k0+1)*16 + ln];
      a.u[j] = lo | (hi << 16);
    }
    v16h bt0 = load_b16(B1 + ((size_t)(cc*2 + 0)*32 + lane)*8);
    v16h bt1 = load_b16(B1 + ((size_t)(cc*2 + 1)*32 + lane)*8);
    acc0 = __builtin_amdgcn_wmma_f32_16x16x32_f16(false, a.h, false, bt0, (short)0, acc0, false, false);
    acc1 = __builtin_amdgcn_wmma_f32_16x16x32_f16(false, a.h, false, bt1, (short)0, acc1, false, false);
  }

  float y0[8], y1[8];
  #pragma unroll
  for (int r = 0; r < 8; ++r) { y0[r] = acc0[r]; y1[r] = acc1[r]; }
  // y / (||y||_D + eps), relu, / (||.||_D + eps); D-rows split across lane^16
  float s0 = 0.f, s1 = 0.f;
  #pragma unroll
  for (int r = 0; r < 8; ++r) { s0 = fmaf(y0[r], y0[r], s0); s1 = fmaf(y1[r], y1[r], s1); }
  s0 += __shfl_xor(s0, 16); s1 += __shfl_xor(s1, 16);
  float i0 = 1.f/(sqrtf(s0)+EPSF), i1 = 1.f/(sqrtf(s1)+EPSF);
  #pragma unroll
  for (int r = 0; r < 8; ++r) { y0[r] = fmaxf(y0[r]*i0, 0.f); y1[r] = fmaxf(y1[r]*i1, 0.f); }
  s0 = 0.f; s1 = 0.f;
  #pragma unroll
  for (int r = 0; r < 8; ++r) { s0 = fmaf(y0[r], y0[r], s0); s1 = fmaf(y1[r], y1[r], s1); }
  s0 += __shfl_xor(s0, 16); s1 += __shfl_xor(s1, 16);
  i0 = 1.f/(sqrtf(s0)+EPSF); i1 = 1.f/(sqrtf(s1)+EPSF);
  #pragma unroll
  for (int r = 0; r < 8; ++r) { y0[r] *= i0; y1[r] *= i1; }

  float p0[8], p1[8];   // partner column (o+1) from lane+1
  #pragma unroll
  for (int r = 0; r < 8; ++r) { p0[r] = __shfl_down(y0[r], 1); p1[r] = __shfl_down(y1[r], 1); }
  if ((ln & 1) == 0) {
    size_t base = (size_t)point * 240;
    #pragma unroll
    for (int r = 0; r < 8; ++r) {
      int dd = r + 8*hf;
      fm1h[base + (size_t)(ln >> 1)*16 + dd] = pack2h(y0[r], p0[r]);          // cols 0..15
      if (ln <= 12)
        fm1h[base + (size_t)(8 + (ln >> 1))*16 + dd] = pack2h(y1[r], p1[r]);  // cols 16..29
    }
  }
}

// ---------------------------------------------------------------------------
// K3: wfm2 + nonlinear. One wave per point, 2 waves/block, 60KB dynamic LDS.
// Stage 32 neighbor rows (240 packed u32 each, contiguous copy) -> LDS tile;
// y[16x40] = A[16x960] x softmax(w2)[960x40] via 30x3 v_wmma_f32_16x16x32_f16.
// ---------------------------------------------------------------------------
__global__ __launch_bounds__(64) void k3_wfm2(const unsigned* __restrict__ fm1h,
                                              const int* __restrict__ idx,
                                              const unsigned* __restrict__ B2,
                                              float* __restrict__ fm2) {
  extern __shared__ char dynsmem[];
  unsigned* ldsall = (unsigned*)dynsmem;
  int wave = threadIdx.x >> 5, lane = threadIdx.x & 31;
  int point = blockIdx.x * 2 + wave;
  int b = point >> 9;
  unsigned* lds = ldsall + wave * 7680;

  int nb = idx[(size_t)point*32 + lane];
  for (int kk = 0; kk < 32; ++kk) {
    int row = __shfl(nb, kk);
    const unsigned* src = fm1h + (size_t)(b*512 + row) * 240;
    for (int off = lane; off < 240; off += 32)
      lds[kk*240 + off] = src[off];
  }
  __syncthreads();

  int hf = lane >> 4, ln = lane & 15;
  v8f acc[3] = {{}, {}, {}};
  for (int cc = 0; cc < 30; ++cc) {
    V16H a;
    #pragma unroll
    for (int j = 0; j < 8; ++j) {
      int kh = ((j < 4) ? j : 8 + (j-4)) + 4*hf;     // (K0>>1)
      a.u[j] = lds[(16*cc + kh)*16 + ln];
    }
    #pragma unroll
    for (int t = 0; t < 3; ++t) {
      v16h bt = load_b16(B2 + ((size_t)(cc*3 + t)*32 + lane)*8);
      acc[t] = __builtin_amdgcn_wmma_f32_16x16x32_f16(false, a.h, false, bt, (short)0, acc[t], false, false);
    }
  }

  #pragma unroll
  for (int t = 0; t < 3; ++t) {
    float y[8];
    #pragma unroll
    for (int r = 0; r < 8; ++r) y[r] = acc[t][r];
    float ss = 0.f;
    #pragma unroll
    for (int r = 0; r < 8; ++r) ss = fmaf(y[r], y[r], ss);
    ss += __shfl_xor(ss, 16);
    float inv = 1.f/(sqrtf(ss)+EPSF);
    #pragma unroll
    for (int r = 0; r < 8; ++r) y[r] = fmaxf(y[r]*inv, 0.f);
    float s2 = 0.f;
    #pragma unroll
    for (int r = 0; r < 8; ++r) s2 = fmaf(y[r], y[r], s2);
    s2 += __shfl_xor(s2, 16);
    float inv2 = 1.f/(sqrtf(s2)+EPSF);
    int o = 16*t + ln;
    if (o < 40) {
      #pragma unroll
      for (int r = 0; r < 8; ++r)
        fm2[(size_t)point*640 + (r + 8*hf)*40 + o] = y[r]*inv2;
    }
  }
}

// ---------------------------------------------------------------------------
// K4a: per-batch mean over N.  K4b: geodesic distance features.
// ---------------------------------------------------------------------------
__global__ __launch_bounds__(640) void k4_mean(const float* __restrict__ fm2,
                                               float* __restrict__ mean) {
  int b = blockIdx.x, t = threadIdx.x;           // t = d*40+c, 640 total
  const float* p = fm2 + (size_t)b*512*640 + t;
  float s = 0.f;
  for (int n = 0; n < 512; ++n) s += p[(size_t)n*640];
  mean[b*640 + t] = s * (1.f/512.f);
}

__global__ __launch_bounds__(256) void k4_feat(const float* __restrict__ fm2,
                                               const float* __restrict__ mean,
                                               float* __restrict__ feat) {
  int g = blockIdx.x*256 + threadIdx.x;          // < 32*512*40
  int c = g % 40, n = (g/40) & 511, b = g/(40*512);
  const float* m = mean + b*640 + c;
  const float* v = fm2 + (size_t)(b*512 + n)*640 + c;
  float msq = 0.f, dot = 0.f;
  #pragma unroll
  for (int d = 0; d < 16; ++d) {
    float mv = m[d*40];
    msq = fmaf(mv, mv, msq);
    dot = fmaf(v[d*40], mv, dot);
  }
  float cosv = dot / (sqrtf(msq) + EPSF);
  cosv = fminf(fmaxf(cosv, -1.f + 1e-6f), 1.f - 1e-6f);
  feat[(size_t)b*20480 + n*40 + c] = acosf(cosv);
}

// ---------------------------------------------------------------------------
// K5: final FC, fp32, one wave per (b, o) output.
// ---------------------------------------------------------------------------
__global__ __launch_bounds__(256) void k5_fc(const float* __restrict__ feat,
                                             const float* __restrict__ fcw,
                                             const float* __restrict__ fcb,
                                             float* __restrict__ out) {
  int wid = blockIdx.x*8 + (threadIdx.x >> 5);
  int lane = threadIdx.x & 31;
  int b = wid / 40, o = wid % 40;
  const float* f = feat + (size_t)b*20480;
  const float* w = fcw  + (size_t)o*20480;
  float acc = 0.f;
  for (int j = lane; j < 20480; j += 32) acc = fmaf(f[j], w[j], acc);
  #pragma unroll
  for (int off = 16; off; off >>= 1) acc += __shfl_down(acc, off);
  if (lane == 0) out[b*40 + o] = acc + fcb[o];
}

// ---------------------------------------------------------------------------
extern "C" void kernel_launch(void* const* d_in, const int* in_sizes, int n_in,
                              void* d_out, int out_size, void* d_ws, size_t ws_size,
                              hipStream_t stream) {
  (void)in_sizes; (void)n_in; (void)out_size; (void)ws_size;
  const float* x   = (const float*)d_in[0];
  const float* w1  = (const float*)d_in[1];
  const float* w2  = (const float*)d_in[2];
  const float* fcw = (const float*)d_in[3];
  const float* fcb = (const float*)d_in[4];
  float* out = (float*)d_out;

  char* ws = (char*)d_ws;
  int*      ws_idx  = (int*)     (ws + 0);            //  2,097,152 B
  unsigned* ws_B1   = (unsigned*)(ws + 2097152);      //      6,144 B
  unsigned* ws_B2   = (unsigned*)(ws + 2103296);      //     92,160 B
  unsigned* ws_fm1h = (unsigned*)(ws + 2195456);      // 15,728,640 B
  float*    ws_fm2  = (float*)   (ws + 17924096);     // 41,943,040 B
  float*    ws_mean = (float*)   (ws + 59867136);     //     81,920 B
  float*    ws_feat = (float*)   (ws + 59949056);     //  2,621,440 B  (total ~62.6 MB)

  k0_prep<<<1, 128, 0, stream>>>(w1, w2, ws_B1, ws_B2);
  k1_knn <<<32, 512, 512*48*sizeof(float), stream>>>(x, ws_idx);
  k2_wfm1<<<2048, 256, 0, stream>>>(x, ws_idx, ws_B1, ws_fm1h);
  k3_wfm2<<<8192, 64, 2*7680*sizeof(unsigned), stream>>>(ws_fm1h, ws_idx, ws_B2, ws_fm2);
  k4_mean<<<32, 640, 0, stream>>>(ws_fm2, ws_mean);
  k4_feat<<<(32*512*40)/256, 256, 0, stream>>>(ws_fm2, ws_mean, ws_feat);
  k5_fc  <<<160, 256, 0, stream>>>(ws_feat, fcw, fcb, out);
}
